// Attention_70188355551857
// MI455X (gfx1250) — matrix operand order repaired and checked by reading, not compiled, not run on hardware
//
#include <hip/hip_runtime.h>

typedef __attribute__((ext_vector_type(16))) __bf16 v16bf;
typedef __attribute__((ext_vector_type(8)))  float  v8f;
typedef __attribute__((ext_vector_type(4)))  unsigned int v4u;

#define WMMA_BF16(a, b, c) \
  __builtin_amdgcn_wmma_f32_16x16x32_bf16(false, (a), false, (b), (short)0, (c), false, false)

__device__ __forceinline__ unsigned short f2bf(float f) {
  unsigned int u = __float_as_uint(f);
  u += 0x7FFFu + ((u >> 16) & 1u);           // round-to-nearest-even
  return (unsigned short)(u >> 16);
}

__device__ __forceinline__ v16bf frag_ld(const unsigned short* p0, const unsigned short* p1) {
  union { v4u u[2]; v16bf b; } c;
  c.u[0] = *(const v4u*)p0;
  c.u[1] = *(const v4u*)p1;
  return c.b;
}

__device__ __forceinline__ v8f zero8() {
  v8f z = {0.f, 0.f, 0.f, 0.f, 0.f, 0.f, 0.f, 0.f};
  return z;
}

// ---------------------------------------------------------------------------
// C = A(f32, MxK) * B(f32, KxN) with bf16 WMMA, f32 accumulate.
// Block tile 128x128, 256 threads (8 waves), wave tile 32x64, K-step 32.
// Double-buffered LDS: global->reg for tile i+1 issued before WMMAs of tile i,
// reg->LDS(buf^1) after compute, single barrier per K-step.
// ---------------------------------------------------------------------------
__global__ __launch_bounds__(256) void gemm_bf16(
    const float* __restrict__ A, const float* __restrict__ B, float* __restrict__ C,
    int M, int N, int K)
{
  __shared__ unsigned short sa[2][128 * 40];   // A tile 128x32 (stride 40)
  __shared__ unsigned short sb[2][128 * 40];   // B tile transposed [n][k] (stride 40)

  const int tid  = threadIdx.x;
  const int wave = tid >> 5, lane = tid & 31;
  const int bm   = blockIdx.y * 128, bn = blockIdx.x * 128;
  const int moff = (wave >> 1) * 32, noff = (wave & 1) * 64;
  const int lrow = lane & 15, lhi = lane >> 4;

  // per-thread staging coordinates (compile-time unrolled)
  int a_row[4], a_c4[4], b_k[4], b_n4[4];
#pragma unroll
  for (int i = 0; i < 4; ++i) {
    int idx = tid + i * 256;
    a_row[i] = idx >> 3; a_c4[i] = (idx & 7) * 4;
    b_k[i]   = idx >> 5; b_n4[i] = (idx & 31) * 4;
  }

  v8f acc[2][4];
#pragma unroll
  for (int i = 0; i < 2; ++i)
#pragma unroll
    for (int j = 0; j < 4; ++j) acc[i][j] = zero8();

  float4 ra[4], rb[4];
  // ---- prologue: stage tile 0 ----
#pragma unroll
  for (int i = 0; i < 4; ++i) {
    ra[i] = *(const float4*)(A + (size_t)(bm + a_row[i]) * K + a_c4[i]);
    rb[i] = *(const float4*)(B + (size_t)(b_k[i]) * N + bn + b_n4[i]);
  }
#pragma unroll
  for (int i = 0; i < 4; ++i) {
    uint2 pk;
    pk.x = f2bf(ra[i].x) | ((unsigned int)f2bf(ra[i].y) << 16);
    pk.y = f2bf(ra[i].z) | ((unsigned int)f2bf(ra[i].w) << 16);
    *(uint2*)(&sa[0][a_row[i] * 40 + a_c4[i]]) = pk;
    sb[0][(b_n4[i] + 0) * 40 + b_k[i]] = f2bf(rb[i].x);
    sb[0][(b_n4[i] + 1) * 40 + b_k[i]] = f2bf(rb[i].y);
    sb[0][(b_n4[i] + 2) * 40 + b_k[i]] = f2bf(rb[i].z);
    sb[0][(b_n4[i] + 3) * 40 + b_k[i]] = f2bf(rb[i].w);
  }
  __syncthreads();

  for (int k0 = 0; k0 < K; k0 += 32) {
    const int  buf  = (k0 >> 5) & 1;
    const bool more = (k0 + 32) < K;

    if (more) {                               // issue next-tile global loads early
      const int kn = k0 + 32;
#pragma unroll
      for (int i = 0; i < 4; ++i) {
        ra[i] = *(const float4*)(A + (size_t)(bm + a_row[i]) * K + kn + a_c4[i]);
        rb[i] = *(const float4*)(B + (size_t)(kn + b_k[i]) * N + bn + b_n4[i]);
      }
      // warm L2 one more tile ahead
      __builtin_prefetch(A + (size_t)(bm + a_row[0]) * K + kn + 32 + a_c4[0], 0, 0);
      __builtin_prefetch(B + (size_t)(kn + 32 + b_k[0]) * N + bn + b_n4[0], 0, 0);
    }

    // ---- compute current tile ----
    v16bf afr[2], bfr[4];
#pragma unroll
    for (int mt = 0; mt < 2; ++mt) {
      const unsigned short* p = &sa[buf][(moff + mt * 16 + lrow) * 40 + lhi * 8];
      afr[mt] = frag_ld(p, p + 16);           // K = {kb..kb+7, kb+16..kb+23}
    }
#pragma unroll
    for (int nt = 0; nt < 4; ++nt) {
      const unsigned short* p = &sb[buf][(noff + nt * 16 + lrow) * 40 + lhi * 8];
      bfr[nt] = frag_ld(p, p + 16);           // K = {kb..kb+7, kb+16..kb+23}
    }
#pragma unroll
    for (int mt = 0; mt < 2; ++mt)
#pragma unroll
      for (int nt = 0; nt < 4; ++nt)
        acc[mt][nt] = WMMA_BF16(afr[mt], bfr[nt], acc[mt][nt]);

    if (more) {                               // store next tile into the other buffer
      const int nb = buf ^ 1;
#pragma unroll
      for (int i = 0; i < 4; ++i) {
        uint2 pk;
        pk.x = f2bf(ra[i].x) | ((unsigned int)f2bf(ra[i].y) << 16);
        pk.y = f2bf(ra[i].z) | ((unsigned int)f2bf(ra[i].w) << 16);
        *(uint2*)(&sa[nb][a_row[i] * 40 + a_c4[i]]) = pk;
        sb[nb][(b_n4[i] + 0) * 40 + b_k[i]] = f2bf(rb[i].x);
        sb[nb][(b_n4[i] + 1) * 40 + b_k[i]] = f2bf(rb[i].y);
        sb[nb][(b_n4[i] + 2) * 40 + b_k[i]] = f2bf(rb[i].z);
        sb[nb][(b_n4[i] + 3) * 40 + b_k[i]] = f2bf(rb[i].w);
      }
      __syncthreads();
    }
  }

#pragma unroll
  for (int mt = 0; mt < 2; ++mt)
#pragma unroll
    for (int nt = 0; nt < 4; ++nt) {
      int col = bn + noff + nt * 16 + lrow;
#pragma unroll
      for (int r = 0; r < 8; ++r) {
        int row = bm + moff + mt * 16 + lhi * 8 + r;
        C[(size_t)row * N + col] = acc[mt][nt][r];
      }
    }
}

// ---------------------------------------------------------------------------
// RoPE (first 1008 positions of q,k) + 1/sqrt(128) scale folded into Q,
// pack f32 qkv -> bf16 Q/K/V in [bc][h][s][d] layout. 4 elements / thread.
// ---------------------------------------------------------------------------
__global__ __launch_bounds__(256) void rope_pack(
    const float* __restrict__ qkv, const float* __restrict__ cosT, const float* __restrict__ sinT,
    unsigned short* __restrict__ Qb, unsigned short* __restrict__ Kb, unsigned short* __restrict__ Vb)
{
  const int idx4 = blockIdx.x * 256 + threadIdx.x;   // 12,582,912 total
  const int hd = (idx4 & 31) * 4;
  int t = idx4 >> 5;
  const int head = t % 48; t /= 48;
  const int s = t & 1023;
  const int bc = t >> 10;

  const size_t src = (((size_t)(bc * 1024 + s)) * 48 + head) * 128 + hd;
  float4 x = *(const float4*)(qkv + src);
  float4 y = x;
  if (head < 32 && s < 1008) {
    const int ph = (hd < 64) ? hd + 64 : hd - 64;
    float4 pr = *(const float4*)(qkv + src - hd + ph);
    const float sgn = (hd < 64) ? -1.f : 1.f;
    float4 c  = *(const float4*)(cosT + s * 128 + hd);
    float4 sn = *(const float4*)(sinT + s * 128 + hd);
    y.x = x.x * c.x + sgn * pr.x * sn.x;
    y.y = x.y * c.y + sgn * pr.y * sn.y;
    y.z = x.z * c.z + sgn * pr.z * sn.z;
    y.w = x.w * c.w + sgn * pr.w * sn.w;
  }
  unsigned short* dst;
  int h;
  if (head < 16) {
    dst = Qb; h = head;
    const float sc = 0.08838834764831845f;   // 1/sqrt(128), folded into Q
    y.x *= sc; y.y *= sc; y.z *= sc; y.w *= sc;
  } else if (head < 32) { dst = Kb; h = head - 16; }
  else                  { dst = Vb; h = head - 32; }

  const size_t o = (((size_t)(bc * 16 + h)) * 1024 + s) * 128 + hd;
  uint2 pk;
  pk.x = f2bf(y.x) | ((unsigned int)f2bf(y.y) << 16);
  pk.y = f2bf(y.z) | ((unsigned int)f2bf(y.w) << 16);
  *(uint2*)(dst + o) = pk;
}

// ---------------------------------------------------------------------------
// Flash attention: 128 threads = 4 waves, each wave owns 16 q-rows.
// 64 keys per iteration: 16 WMMAs for S = Q*K^T (Q resident in A-frags),
// online softmax (intra-half shuffles, amortized over 64 keys),
// P via per-wave LDS -> A-frags, 16 WMMAs for P*V.
// Writes fp32 attn in [bc*s][h*128+d] layout for the O-proj GEMM.
// ---------------------------------------------------------------------------
__global__ __launch_bounds__(128) void flash_attn(
    const unsigned short* __restrict__ Qb, const unsigned short* __restrict__ Kb,
    const unsigned short* __restrict__ Vb, float* __restrict__ attn)
{
  __shared__ unsigned short sk[64 * 136];    // K block [key][d], stride 136
  __shared__ unsigned short sv[128 * 72];    // V block transposed [d][key], stride 72
  __shared__ unsigned short sp[4][16 * 72];  // per-wave P tile [row][key], stride 72

  const int tid  = threadIdx.x;
  const int wave = tid >> 5, lane = tid & 31;
  const int lrow = lane & 15, lhi = lane >> 4;
  const int bh = blockIdx.x;                 // bc*16 + h
  const int bc = bh >> 4, h = bh & 15;
  const int q0 = blockIdx.y * 64 + wave * 16;
  const size_t base = (size_t)bh * 1024 * 128;

  // Q fragments resident (16 rows x 128, 4 K-chunks of 32)
  v16bf qf[4];
#pragma unroll
  for (int kc = 0; kc < 4; ++kc) {
    const unsigned short* p = Qb + base + (size_t)(q0 + lrow) * 128 + kc * 32 + lhi * 8;
    qf[kc] = frag_ld(p, p + 16);
  }

  float m[8], l[8];
  v8f acc[8];
#pragma unroll
  for (int r = 0; r < 8; ++r) { m[r] = -1e30f; l[r] = 0.f; }
#pragma unroll
  for (int t2 = 0; t2 < 8; ++t2) acc[t2] = zero8();

  for (int j0 = 0; j0 < 1024; j0 += 64) {
    __syncthreads();
#pragma unroll
    for (int i = 0; i < 8; ++i) {
      int idx = tid + i * 128;               // 0..1023 uint4 slots
      int key = idx >> 4, h8 = (idx & 15) * 8;
      v4u kv = *(const v4u*)(Kb + base + (size_t)(j0 + key) * 128 + h8);
      *(v4u*)(&sk[key * 136 + h8]) = kv;
      v4u vv = *(const v4u*)(Vb + base + (size_t)(j0 + key) * 128 + h8);
#pragma unroll
      for (int e = 0; e < 4; ++e) {
        unsigned int w = vv[e];
        sv[(h8 + 2 * e + 0) * 72 + key] = (unsigned short)(w & 0xFFFFu);
        sv[(h8 + 2 * e + 1) * 72 + key] = (unsigned short)(w >> 16);
      }
    }
    if (j0 + 64 < 1024) {                    // warm L2 for the next key block
      __builtin_prefetch(Kb + base + (size_t)(j0 + 64 + (tid >> 1)) * 128 + (tid & 1) * 64, 0, 0);
      __builtin_prefetch(Vb + base + (size_t)(j0 + 64 + (tid >> 1)) * 128 + (tid & 1) * 64, 0, 0);
    }
    __syncthreads();

    // scores for 4 key sub-tiles of 16
    v8f sc[4];
#pragma unroll
    for (int nt = 0; nt < 4; ++nt) sc[nt] = zero8();
#pragma unroll
    for (int nt = 0; nt < 4; ++nt)
#pragma unroll
      for (int kc = 0; kc < 4; ++kc) {
        const unsigned short* p = &sk[(nt * 16 + lrow) * 136 + kc * 32 + lhi * 8];
        sc[nt] = WMMA_BF16(qf[kc], frag_ld(p, p + 16), sc[nt]);
      }

    // online softmax, row r lives in lane half lhi
#pragma unroll
    for (int r = 0; r < 8; ++r) {
      float tm = fmaxf(fmaxf(sc[0][r], sc[1][r]), fmaxf(sc[2][r], sc[3][r]));
#pragma unroll
      for (int d = 1; d < 16; d <<= 1) tm = fmaxf(tm, __shfl_xor(tm, d, 32));
      float mn   = fmaxf(m[r], tm);
      float corr = __expf(m[r] - mn);
      float rs = 0.f;
#pragma unroll
      for (int nt = 0; nt < 4; ++nt) {
        float p = __expf(sc[nt][r] - mn);
        sc[nt][r] = p;
        rs += p;
      }
#pragma unroll
      for (int d = 1; d < 16; d <<= 1) rs += __shfl_xor(rs, d, 32);
      l[r] = l[r] * corr + rs;
      m[r] = mn;
#pragma unroll
      for (int t2 = 0; t2 < 8; ++t2) acc[t2][r] *= corr;
    }

    // P (C-layout) -> per-wave LDS -> A-fragment layout (same-wave DS is in-order)
    unsigned short* pw = sp[wave];
#pragma unroll
    for (int r = 0; r < 8; ++r) {
      int row = r + lhi * 8;
#pragma unroll
      for (int nt = 0; nt < 4; ++nt)
        pw[row * 72 + nt * 16 + lrow] = f2bf(sc[nt][r]);
    }
    v16bf pf[2];
#pragma unroll
    for (int kc = 0; kc < 2; ++kc) {
      const unsigned short* pp = &pw[lrow * 72 + kc * 32 + lhi * 8];
      pf[kc] = frag_ld(pp, pp + 16);
    }

    // O += P * V  (8 hd sub-tiles x 2 key chunks)
#pragma unroll
    for (int t2 = 0; t2 < 8; ++t2)
#pragma unroll
      for (int kc = 0; kc < 2; ++kc) {
        const unsigned short* vp = &sv[(t2 * 16 + lrow) * 72 + kc * 32 + lhi * 8];
        acc[t2] = WMMA_BF16(pf[kc], frag_ld(vp, vp + 16), acc[t2]);
      }
  }

#pragma unroll
  for (int r = 0; r < 8; ++r) {
    float inv = 1.0f / l[r];
#pragma unroll
    for (int t2 = 0; t2 < 8; ++t2) acc[t2][r] *= inv;
  }
#pragma unroll
  for (int t2 = 0; t2 < 8; ++t2) {
    int col = h * 128 + t2 * 16 + lrow;
#pragma unroll
    for (int r = 0; r < 8; ++r) {
      int row = bc * 1024 + q0 + lhi * 8 + r;
      attn[(size_t)row * 2048 + col] = acc[t2][r];
    }
  }
}

// ---------------------------------------------------------------------------
extern "C" void kernel_launch(void* const* d_in, const int* in_sizes, int n_in,
                              void* d_out, int out_size, void* d_ws, size_t ws_size,
                              hipStream_t stream) {
  (void)in_sizes; (void)n_in; (void)out_size; (void)ws_size;
  const float* hidden = (const float*)d_in[0];   // (2,4,1024,2048)
  const float* cosT   = (const float*)d_in[1];   // (1008,128)
  const float* sinT   = (const float*)d_in[2];   // (1008,128)
  const float* w_qkv  = (const float*)d_in[3];   // (2048,6144)
  const float* w_o    = (const float*)d_in[4];   // (2048,2048)
  float* out = (float*)d_out;

  char* ws = (char*)d_ws;
  float*          qkv  = (float*)ws;                                 // 8192*6144*4   = 201326592 B
  unsigned short* Qb   = (unsigned short*)(ws + 201326592);          // 8*16*1024*128*2 = 33554432 B
  unsigned short* Kb   = (unsigned short*)(ws + 234881024);
  unsigned short* Vb   = (unsigned short*)(ws + 268435456);
  float*          attn = (float*)(ws + 301989888);                   // 8192*2048*4

  // 1) QKV projection: (8192x2048) * (2048x6144)
  dim3 g1(6144 / 128, 8192 / 128);
  gemm_bf16<<<g1, 256, 0, stream>>>(hidden, w_qkv, qkv, 8192, 6144, 2048);

  // 2) RoPE + bf16 pack
  rope_pack<<<49152, 256, 0, stream>>>(qkv, cosT, sinT, Qb, Kb, Vb);

  // 3) Flash attention: 8 bc * 16 heads, 16 q-blocks of 64 rows
  dim3 g3(128, 16);
  flash_attn<<<g3, 128, 0, stream>>>(Qb, Kb, Vb, attn);

  // 4) Output projection: (8192x2048) * (2048x2048)
  dim3 g4(2048 / 128, 8192 / 128);
  gemm_bf16<<<g4, 256, 0, stream>>>(attn, w_o, out, 8192, 2048, 2048);
}
